// RGBuvHistBlock_67551245632192
// MI455X (gfx1250) — matrix operand order repaired and checked by reading, not compiled, not run on hardware
//
#include <hip/hip_runtime.h>
#include <hip/hip_bf16.h>
#include <math.h>

typedef __attribute__((ext_vector_type(16))) _Float16 v16h;
typedef __attribute__((ext_vector_type(8)))  _Float16 v8h;
typedef __attribute__((ext_vector_type(2)))  __fp16   v2hf;   // matches cvt_pkrtz return type
typedef __attribute__((ext_vector_type(8)))  float    v8f;

#define NPIX     65536        // 256*256 pixels per plane
#define NBINS    64
#define EPS_F    1e-6f
#define P_WGS    32           // workgroups per (b,c) pair
#define CHUNK    128          // pixels per LDS stage (4 WMMA K-steps)
#define ROWL     136          // padded LDS row (halves): 272B rows, 16B-aligned frags
#define LN2x50   34.6573590280f   // ln(2) * (1/sigma) = 0.693147 * 50
#define SQ_INVS  50.0f            // 1/sigma

// ---------------------------------------------------------------- zero output
__global__ void hist_zero_kernel(float* __restrict__ out, int n) {
  int i = blockIdx.x * blockDim.x + threadIdx.x;
  if (i < n) out[i] = 0.0f;
}

// ---------------------------------------------------------------- histogram GEMM
// grid.x = 12 * P_WGS, block = 128 threads (4 waves)
__global__ __launch_bounds__(128) void hist_wmma_kernel(const float* __restrict__ x,
                                                        float* __restrict__ out) {
  __shared__ __align__(16) _Float16 Alds[NBINS][ROWL];  // a = Iy*Ku, [u][k]
  __shared__ __align__(16) _Float16 Blds[NBINS][ROWL];  // Kv,        [v][k]

  const int bc   = blockIdx.x % 12;
  const int part = blockIdx.x / 12;
  const int b    = bc / 3;
  const int c    = bc % 3;

  const int tid  = threadIdx.x;
  const int pp   = tid & 63;         // pixel pair id: pixels 2pp, 2pp+1 of chunk
  const int bh   = tid >> 6;         // bin half: bins [32*bh, 32*bh+32)
  const int wv   = tid >> 5;         // wave id == v-tile id
  const int lane = tid & 31;

  const float* xb = x + (size_t)b * 3 * NPIX;

  v8f acc0 = {}, acc1 = {}, acc2 = {}, acc3 = {};

  const int pxPerWG = NPIX / P_WGS;          // 2048 -> 16 chunks of 128
  const int base    = part * pxPerWG;

  // fragment addressing (constant per lane)
  const int fm   = lane & 15;                // A row within tile / B col within tile
  const int fkh  = (lane >> 4) * 8;          // A: K half-offset
  const int fks  = (lane >> 4) * 16;         // unused: B also uses fkh, same k order as A
  const int binBase = bh * 32;

  for (int ch = 0; ch < pxPerWG; ch += CHUNK) {
    // ---- cooperative tile build: each thread = 2 pixels x 32 bins
    {
      const int n = base + ch + 2 * pp;
      float2 r2 = *(const float2*)&xb[n];
      float2 g2 = *(const float2*)&xb[NPIX + n];
      float2 b2 = *(const float2*)&xb[2 * NPIX + n];
      const float r0 = fminf(fmaxf(r2.x, 0.0f), 1.0f), r1 = fminf(fmaxf(r2.y, 0.0f), 1.0f);
      const float g0 = fminf(fmaxf(g2.x, 0.0f), 1.0f), g1 = fminf(fmaxf(g2.y, 0.0f), 1.0f);
      const float u0 = fminf(fmaxf(b2.x, 0.0f), 1.0f), u1 = fminf(fmaxf(b2.y, 0.0f), 1.0f);
      const float iy0 = __builtin_amdgcn_sqrtf(r0*r0 + g0*g0 + u0*u0 + EPS_F);
      const float iy1 = __builtin_amdgcn_sqrtf(r1*r1 + g1*g1 + u1*u1 + EPS_F);
      // log2 of channels (inputs >= EPS, raw v_log_f32 is safe)
      const float lr0 = __builtin_amdgcn_logf(r0 + EPS_F), lr1 = __builtin_amdgcn_logf(r1 + EPS_F);
      const float lg0 = __builtin_amdgcn_logf(g0 + EPS_F), lg1 = __builtin_amdgcn_logf(g1 + EPS_F);
      const float lb0 = __builtin_amdgcn_logf(u0 + EPS_F), lb1 = __builtin_amdgcn_logf(u1 + EPS_F);
      // pre-scaled chroma coords: Us = (ln(a)-ln(b))/sigma = (log2 a - log2 b)*ln2*50
      float Us0, Us1, Vs0, Vs1;
      if      (c == 0) { Us0 = (lr0-lg0)*LN2x50; Us1 = (lr1-lg1)*LN2x50;
                         Vs0 = (lr0-lb0)*LN2x50; Vs1 = (lr1-lb1)*LN2x50; }
      else if (c == 1) { Us0 = (lg0-lr0)*LN2x50; Us1 = (lg1-lr1)*LN2x50;
                         Vs0 = (lg0-lb0)*LN2x50; Vs1 = (lg1-lb1)*LN2x50; }
      else             { Us0 = (lb0-lr0)*LN2x50; Us1 = (lb1-lr1)*LN2x50;
                         Vs0 = (lb0-lg0)*LN2x50; Vs1 = (lb1-lg1)*LN2x50; }

      #pragma unroll 4
      for (int i = 0; i < 32; ++i) {
        const int   bin = binBase + i;
        const float bs  = (-3.0f + (float)bin * (6.0f / 63.0f)) * SQ_INVS;
        const float d0 = Us0 - bs, d1 = Us1 - bs;
        const float e0 = Vs0 - bs, e1 = Vs1 - bs;
        const float a0 = iy0 * __builtin_amdgcn_rcpf(__builtin_fmaf(d0, d0, 1.0f));
        const float a1 = iy1 * __builtin_amdgcn_rcpf(__builtin_fmaf(d1, d1, 1.0f));
        const float k0 =       __builtin_amdgcn_rcpf(__builtin_fmaf(e0, e0, 1.0f));
        const float k1 =       __builtin_amdgcn_rcpf(__builtin_fmaf(e1, e1, 1.0f));
        *(v2hf*)&Alds[bin][2 * pp] = __builtin_amdgcn_cvt_pkrtz(a0, a1);
        *(v2hf*)&Blds[bin][2 * pp] = __builtin_amdgcn_cvt_pkrtz(k0, k1);
      }
    }
    __syncthreads();

    // ---- 4 K-steps x 4 u-tiles = 16 WMMAs per wave per stage.
    // Preload all 10 independent b128 fragments per K-step so a single
    // dscnt wait covers the clause and the 4 WMMAs issue back-to-back.
    union U16 { v16h v; v8h h[2]; };
    #pragma unroll
    for (int kb = 0; kb < CHUNK; kb += 32) {
      U16 bf, af0, af1, af2, af3;
      bf.h[0]  = *(const v8h*)&Blds[wv * 16 + fm][kb + fkh];
      bf.h[1]  = *(const v8h*)&Blds[wv * 16 + fm][kb + 16 + fkh];
      af0.h[0] = *(const v8h*)&Alds[ 0 + fm][kb + fkh];
      af0.h[1] = *(const v8h*)&Alds[ 0 + fm][kb + 16 + fkh];
      af1.h[0] = *(const v8h*)&Alds[16 + fm][kb + fkh];
      af1.h[1] = *(const v8h*)&Alds[16 + fm][kb + 16 + fkh];
      af2.h[0] = *(const v8h*)&Alds[32 + fm][kb + fkh];
      af2.h[1] = *(const v8h*)&Alds[32 + fm][kb + 16 + fkh];
      af3.h[0] = *(const v8h*)&Alds[48 + fm][kb + fkh];
      af3.h[1] = *(const v8h*)&Alds[48 + fm][kb + 16 + fkh];
      acc0 = __builtin_amdgcn_wmma_f32_16x16x32_f16(false, af0.v, false, bf.v,
                                                    (short)0, acc0, false, false);
      acc1 = __builtin_amdgcn_wmma_f32_16x16x32_f16(false, af1.v, false, bf.v,
                                                    (short)0, acc1, false, false);
      acc2 = __builtin_amdgcn_wmma_f32_16x16x32_f16(false, af2.v, false, bf.v,
                                                    (short)0, acc2, false, false);
      acc3 = __builtin_amdgcn_wmma_f32_16x16x32_f16(false, af3.v, false, bf.v,
                                                    (short)0, acc3, false, false);
    }
    __syncthreads();
  }

  // ---- combine partial 64x64 hists: C/D layout lane L, vgpr r -> (M=r+8*(L/16), N=L%16)
  float* hb = out + (size_t)(b * 3 + c) * NBINS * NBINS;
  const int vcol = wv * 16 + (lane & 15);
  const int rOff = (lane >> 4) * 8;
  #pragma unroll
  for (int r = 0; r < 8; ++r) {
    atomicAdd(&hb[( 0 + r + rOff) * NBINS + vcol], acc0[r]);
    atomicAdd(&hb[(16 + r + rOff) * NBINS + vcol], acc1[r]);
    atomicAdd(&hb[(32 + r + rOff) * NBINS + vcol], acc2[r]);
    atomicAdd(&hb[(48 + r + rOff) * NBINS + vcol], acc3[r]);
  }
}

// ---------------------------------------------------------------- per-batch normalize
__global__ __launch_bounds__(256) void hist_norm_kernel(float* __restrict__ out) {
  __shared__ float red[256];
  const int bsz = 3 * NBINS * NBINS;          // 12288 per batch
  float* hb = out + (size_t)blockIdx.x * bsz;
  float s = 0.0f;
  for (int i = threadIdx.x; i < bsz; i += 256) s += hb[i];
  red[threadIdx.x] = s;
  __syncthreads();
  for (int off = 128; off > 0; off >>= 1) {
    if (threadIdx.x < off) red[threadIdx.x] += red[threadIdx.x + off];
    __syncthreads();
  }
  const float inv = 1.0f / (red[0] + EPS_F);
  for (int i = threadIdx.x; i < bsz; i += 256) hb[i] *= inv;
}

extern "C" void kernel_launch(void* const* d_in, const int* in_sizes, int n_in,
                              void* d_out, int out_size, void* d_ws, size_t ws_size,
                              hipStream_t stream) {
  (void)in_sizes; (void)n_in; (void)d_ws; (void)ws_size;
  const float* x  = (const float*)d_in[0];
  float* out      = (float*)d_out;

  // 1) zero the accumulator (harness does not re-zero between replays)
  hist_zero_kernel<<<(out_size + 255) / 256, 256, 0, stream>>>(out, out_size);

  // 2) WMMA outer-product histograms: 12 (b,c) pairs x P_WGS partial workgroups
  hist_wmma_kernel<<<12 * P_WGS, 128, 0, stream>>>(x, out);

  // 3) per-batch normalization (one block per batch, in place)
  hist_norm_kernel<<<4, 256, 0, stream>>>(out);
}
